// SparseRPN_18726057410608
// MI455X (gfx1250) — matrix operand order repaired and checked by reading, not compiled, not run on hardware
//
#include <hip/hip_runtime.h>
#include <hip/hip_bf16.h>

typedef __attribute__((ext_vector_type(16))) _Float16 v16h;
typedef __attribute__((ext_vector_type(8)))  _Float16 h8;
typedef __attribute__((ext_vector_type(8)))  float    v8f;

#define BN_EPS 1e-3f

// ---------------------------------------------------------------- utility
__global__ void zero_f32_kernel(float* __restrict__ p, long n) {
    long i = blockIdx.x * (long)blockDim.x + threadIdx.x;
    if (i < n) p[i] = 0.f;
}

__global__ void scatter_kernel(const float* __restrict__ vf, const int* __restrict__ coors,
                               float* __restrict__ x0, float* __restrict__ m0,
                               int nvox, int G) {
    long i = blockIdx.x * (long)blockDim.x + threadIdx.x;
    if (i >= (long)nvox * 64) return;
    int v = (int)(i >> 6), c = (int)(i & 63);
    int b = coors[v * 4 + 0];
    int y = coors[v * 4 + 2];
    int x = coors[v * 4 + 3];
    size_t pix = ((size_t)b * G + y) * G + x;
    atomicAdd(&x0[pix * 64 + c], vf[i]);
    if (c == 0) m0[pix] = 1.0f;
}

__global__ void cvt_f16_kernel(const float* __restrict__ in, _Float16* __restrict__ out, long n) {
    long i = blockIdx.x * (long)blockDim.x + threadIdx.x;
    if (i < n) out[i] = (_Float16)in[i];
}

__global__ void pool_mask_kernel(const float* __restrict__ mi, float* __restrict__ mo,
                                 int B_, int Ho, int Wo) {
    long i = blockIdx.x * (long)blockDim.x + threadIdx.x;
    long tot = (long)B_ * Ho * Wo;
    if (i >= tot) return;
    int b = (int)(i / ((long)Ho * Wo));
    int r = (int)(i % ((long)Ho * Wo));
    int y = r / Wo, x = r % Wo;
    const float* p = mi + (((size_t)b * (2 * Ho) + 2 * y) * (2 * Wo) + 2 * x);
    float v = fmaxf(fmaxf(p[0], p[1]), fmaxf(p[2 * Wo], p[2 * Wo + 1]));
    mo[i] = v;
}

__global__ void mask_count_kernel(const float* __restrict__ m, long n, float scale,
                                  float* __restrict__ out) {
    __shared__ float sh[256];
    float s = 0.f;
    for (long i = threadIdx.x; i < n; i += 256) s += m[i];
    sh[threadIdx.x] = s;
    __syncthreads();
    for (int o = 128; o > 0; o >>= 1) {
        if ((int)threadIdx.x < o) sh[threadIdx.x] += sh[threadIdx.x + o];
        __syncthreads();
    }
    if (threadIdx.x == 0) out[0] = fmaxf(sh[0] * scale, 1.0f);
}

// w layout: [taps][Cin][Cout] f32 (HWIO flattened).
// dec==0: wT[co][tap*Cin+ci]   (GEMM N=Cout, K=taps*Cin)
// dec==1: wT[tap*Cout+co][ci]  (deconv GEMM N=taps*Cout, K=Cin)
__global__ void repack_kernel(const float* __restrict__ w, _Float16* __restrict__ wT,
                              int taps, int Cin, int Cout, int dec) {
    long i = blockIdx.x * (long)blockDim.x + threadIdx.x;
    long tot = (long)taps * Cin * Cout;
    if (i >= tot) return;
    int tap = (int)(i / ((long)Cin * Cout));
    long r = i % ((long)Cin * Cout);
    int ci = (int)(r / Cout), co = (int)(r % Cout);
    _Float16 v = (_Float16)w[i];
    if (!dec)
        wT[(size_t)co * ((size_t)taps * Cin) + (size_t)tap * Cin + ci] = v;
    else
        wT[((size_t)tap * Cout + co) * (size_t)Cin + ci] = v;
}

// ------------------------------------------------------ WMMA implicit GEMM
// Batch is fixed at 2 (reference B=2), so the batch split is a compare.
// One wave computes a 16(M) x 64(N) tile: 4 accumulators sharing one A frag.
// blockIdx.x*8+wave -> M tile (16 consecutive pixels, always one row since
// W % 16 == 0); blockIdx.y -> 64-wide N block.
// in:  f16 NHWC [2][Hin][Win][Cin]
// wT:  f16 [Nout][KH*KW*Cin]
// out: f32; ups==1 -> [2][Hg][Wg][Nout]
//      ups>1  -> deconv scatter to [2][Hg*ups][Wg*ups][CoutReal],
//                n = tap*CoutReal+co, tap = di*ups+dj
__global__ __launch_bounds__(256) void conv_wmma_kernel(
    const _Float16* __restrict__ in, const _Float16* __restrict__ wT,
    float* __restrict__ out,
    int Hin, int Win, int Cin,
    int Hg, int Wg, int Nout,
    int KH, int KW, int stride, int pad,
    int ups, int CoutReal, int mtiles) {
    const int lane = threadIdx.x & 31;
    const int wave = threadIdx.x >> 5;
    int mt = blockIdx.x * 8 + wave;
    if (mt >= mtiles) return;

    const int hw = Hg * Wg;
    const int sel = lane >> 4;   // K-striping half of the wave
    const int ln = lane & 15;

    // row coordinates of this 16-pixel M tile (computed once)
    int gm0 = mt << 4;
    int b0 = (gm0 >= hw) ? 1 : 0;
    int r00 = gm0 - b0 * hw;
    int oy0 = r00 / Wg;
    int ox0 = r00 - oy0 * Wg;
    int ox = ox0 + ln;           // this lane's A-row pixel (x)

    const int Ktot = KH * KW * Cin;
    const int nb = blockIdx.y << 6;
    const _Float16* w0 = wT + (size_t)(nb + ln) * Ktot + sel * 16;
    const size_t wstep = (size_t)16 * Ktot;

    v8f acc0 = {}, acc1 = {}, acc2 = {}, acc3 = {};

    for (int kh = 0; kh < KH; ++kh) {
        int iy = oy0 * stride - pad + kh;
        bool vy = (iy >= 0) && (iy < Hin);
        int siy = vy ? iy : 0;
        for (int kw = 0; kw < KW; ++kw) {
            int ix = ox * stride - pad + kw;
            bool valid = vy && (ix >= 0) && (ix < Win);
            int six = valid ? ix : 0;
            const _Float16* abase = in + ((size_t)(b0 * Hin + siy) * Win + six) * Cin;
            int ko = (kh * KW + kw) * Cin;
            for (int cb = 0; cb < Cin; cb += 32) {
                h8 a0 = {}, a1 = {};
                if (valid) {
                    a0 = *(const h8*)(abase + cb + sel * 16);
                    a1 = *(const h8*)(abase + cb + sel * 16 + 8);
                }
                v16h A = __builtin_shufflevector(a0, a1, 0, 1, 2, 3, 4, 5, 6, 7,
                                                 8, 9, 10, 11, 12, 13, 14, 15);
                const _Float16* wp = w0 + ko + cb;
                {
                    h8 q0 = *(const h8*)(wp);
                    h8 q1 = *(const h8*)(wp + 8);
                    v16h Bv = __builtin_shufflevector(q0, q1, 0, 1, 2, 3, 4, 5, 6, 7,
                                                      8, 9, 10, 11, 12, 13, 14, 15);
                    acc0 = __builtin_amdgcn_wmma_f32_16x16x32_f16(
                        false, A, false, Bv, (short)0, acc0, false, false);
                }
                wp += wstep;
                {
                    h8 q0 = *(const h8*)(wp);
                    h8 q1 = *(const h8*)(wp + 8);
                    v16h Bv = __builtin_shufflevector(q0, q1, 0, 1, 2, 3, 4, 5, 6, 7,
                                                      8, 9, 10, 11, 12, 13, 14, 15);
                    acc1 = __builtin_amdgcn_wmma_f32_16x16x32_f16(
                        false, A, false, Bv, (short)0, acc1, false, false);
                }
                wp += wstep;
                {
                    h8 q0 = *(const h8*)(wp);
                    h8 q1 = *(const h8*)(wp + 8);
                    v16h Bv = __builtin_shufflevector(q0, q1, 0, 1, 2, 3, 4, 5, 6, 7,
                                                      8, 9, 10, 11, 12, 13, 14, 15);
                    acc2 = __builtin_amdgcn_wmma_f32_16x16x32_f16(
                        false, A, false, Bv, (short)0, acc2, false, false);
                }
                wp += wstep;
                {
                    h8 q0 = *(const h8*)(wp);
                    h8 q1 = *(const h8*)(wp + 8);
                    v16h Bv = __builtin_shufflevector(q0, q1, 0, 1, 2, 3, 4, 5, 6, 7,
                                                      8, 9, 10, 11, 12, 13, 14, 15);
                    acc3 = __builtin_amdgcn_wmma_f32_16x16x32_f16(
                        false, A, false, Bv, (short)0, acc3, false, false);
                }
            }
        }
    }

    // store: D tile j covers N columns nb+16j .. nb+16j+15; lane ln -> col,
    // VGPR rr -> row rr+8*sel; all 16 rows share oy0 (single image row).
    v8f accs[4] = {acc0, acc1, acc2, acc3};
#pragma unroll
    for (int j = 0; j < 4; ++j) {
        int nA = nb + (j << 4) + ln;
        if (ups == 1) {
            float* ob = out + ((size_t)(b0 * Hg + oy0) * Wg + ox0) * Nout + nA;
#pragma unroll
            for (int rr = 0; rr < 8; ++rr) {
                int xo = rr + 8 * sel;
                ob[(size_t)xo * Nout] = accs[j][rr];
            }
        } else {
            int tap = nA / CoutReal;
            int co = nA - tap * CoutReal;
            int di = tap / ups, dj = tap % ups;
            int Ho = Hg * ups, Wo = Wg * ups;
            float* ob = out + ((size_t)(b0 * Ho + oy0 * ups + di) * Wo) * CoutReal + co;
#pragma unroll
            for (int rr = 0; rr < 8; ++rr) {
                int xx = ox0 + rr + 8 * sel;
                ob[(size_t)(xx * ups + dj) * CoutReal] = accs[j][rr];
            }
        }
    }
}

// ------------------------------------------------------------- masked BN
__global__ void bn_stats_kernel(const float* __restrict__ y, const float* __restrict__ mask,
                                float* __restrict__ stats, int B_, int Hg, int Wg, int C, int ms) {
    __shared__ float sh1[256], sh2[256];
    int c = blockIdx.x;
    int hw = Hg * Wg;
    int Hm = Hg / ms, Wm = Wg / ms;
    long tot = (long)B_ * hw;
    float s1 = 0.f, s2 = 0.f;
    for (long p = threadIdx.x; p < tot; p += 256) {
        int b = (int)(p / hw);
        int r = (int)(p % hw);
        int yy = r / Wg, xx = r % Wg;
        float m = mask[((size_t)b * Hm + yy / ms) * Wm + xx / ms];
        float v = y[p * C + c];
        s1 += v * m;
        s2 += v * v * m;
    }
    sh1[threadIdx.x] = s1;
    sh2[threadIdx.x] = s2;
    __syncthreads();
    for (int o = 128; o > 0; o >>= 1) {
        if ((int)threadIdx.x < o) {
            sh1[threadIdx.x] += sh1[threadIdx.x + o];
            sh2[threadIdx.x] += sh2[threadIdx.x + o];
        }
        __syncthreads();
    }
    if (threadIdx.x == 0) {
        stats[2 * c] = sh1[0];
        stats[2 * c + 1] = sh2[0];
    }
}

__global__ void bn_finalize_kernel(const float* __restrict__ stats, const float* __restrict__ cnt,
                                   const float* __restrict__ g, const float* __restrict__ b,
                                   float* __restrict__ coef, int C) {
    int c = threadIdx.x;
    if (c >= C) return;
    float n = cnt[0];
    float mean = stats[2 * c] / n;
    float var = stats[2 * c + 1] / n - mean * mean;
    float a = rsqrtf(var + BN_EPS) * g[c];
    coef[2 * c] = a;
    coef[2 * c + 1] = b[c] - mean * a;
}

__global__ void bn_apply_kernel(const float* __restrict__ y, const float* __restrict__ mask,
                                const float* __restrict__ coef, _Float16* __restrict__ out,
                                int B_, int Hg, int Wg, int C, int ms) {
    long i = blockIdx.x * (long)blockDim.x + threadIdx.x;
    long tot = (long)B_ * Hg * Wg * C;
    if (i >= tot) return;
    int c = (int)(i % C);
    long p = i / C;
    int hw = Hg * Wg;
    int b = (int)(p / hw);
    int r = (int)(p % hw);
    int yy = r / Wg, xx = r % Wg;
    int Hm = Hg / ms, Wm = Wg / ms;
    float m = mask[((size_t)b * Hm + yy / ms) * Wm + xx / ms];
    float v = y[i] * coef[2 * c] + coef[2 * c + 1];
    v = fmaxf(v, 0.f) * m;
    out[i] = (_Float16)v;
}

// ----------------------------------------------------------------- heads
__global__ void heads_kernel(const _Float16* __restrict__ up1, const _Float16* __restrict__ up2,
                             const _Float16* __restrict__ up3,
                             const float* __restrict__ bw, const float* __restrict__ bbias,
                             const float* __restrict__ cw, const float* __restrict__ cbias,
                             const float* __restrict__ dw, const float* __restrict__ dbias,
                             float* __restrict__ out, int P) {
    long i = blockIdx.x * (long)blockDim.x + threadIdx.x;
    if (i >= (long)P * 22) return;
    int p = (int)(i / 22), oc = (int)(i % 22);
    const float* w;
    float bias;
    float* dst;
    int OC, k;
    if (oc < 14) {
        k = oc; w = bw; bias = bbias[k]; OC = 14;
        dst = out + (size_t)p * 14 + k;
    } else if (oc < 18) {
        k = oc - 14; w = cw; bias = cbias[k]; OC = 4;
        dst = out + (size_t)P * 14 + (size_t)p * 4 + k;
    } else {
        k = oc - 18; w = dw; bias = dbias[k]; OC = 4;
        dst = out + (size_t)P * 18 + (size_t)p * 4 + k;
    }
    const _Float16* u1 = up1 + (size_t)p * 64;
    const _Float16* u2 = up2 + (size_t)p * 64;
    const _Float16* u3 = up3 + (size_t)p * 128;
    float s = bias;
    for (int ci = 0; ci < 64; ++ci) s += (float)u1[ci] * w[ci * OC + k];
    for (int ci = 0; ci < 64; ++ci) s += (float)u2[ci] * w[(64 + ci) * OC + k];
    for (int ci = 0; ci < 128; ++ci) s += (float)u3[ci] * w[(128 + ci) * OC + k];
    *dst = s;
}

// ---------------------------------------------------------------- driver
extern "C" void kernel_launch(void* const* d_in, const int* in_sizes, int n_in,
                              void* d_out, int out_size, void* d_ws, size_t ws_size,
                              hipStream_t stream) {
    const float* vf   = (const float*)d_in[0];
    const int*   coors = (const int*)d_in[1];
    const float* w1 = (const float*)d_in[3];
    const float* g1 = (const float*)d_in[4];
    const float* b1 = (const float*)d_in[5];
    const float* s1w = (const float*)d_in[6];
    const float* s1g = (const float*)d_in[7];
    const float* s1b = (const float*)d_in[8];
    const float* w2 = (const float*)d_in[9];
    const float* g2 = (const float*)d_in[10];
    const float* b2 = (const float*)d_in[11];
    const float* s2w = (const float*)d_in[12];
    const float* s2g = (const float*)d_in[13];
    const float* s2b = (const float*)d_in[14];
    const float* w3 = (const float*)d_in[15];
    const float* g3 = (const float*)d_in[16];
    const float* b3 = (const float*)d_in[17];
    const float* s3w = (const float*)d_in[18];
    const float* s3g = (const float*)d_in[19];
    const float* s3b = (const float*)d_in[20];
    const float* d1w = (const float*)d_in[21];
    const float* d1g = (const float*)d_in[22];
    const float* d1b = (const float*)d_in[23];
    const float* d2w = (const float*)d_in[24];
    const float* d2g = (const float*)d_in[25];
    const float* d2b = (const float*)d_in[26];
    const float* d3w = (const float*)d_in[27];
    const float* d3g = (const float*)d_in[28];
    const float* d3b = (const float*)d_in[29];
    const float* boxw = (const float*)d_in[30];
    const float* boxb = (const float*)d_in[31];
    const float* clsw = (const float*)d_in[32];
    const float* clsb = (const float*)d_in[33];
    const float* dirw = (const float*)d_in[34];
    const float* dirb = (const float*)d_in[35];

    const int Bn = 2, G = 384, H1 = 192, H2 = 96, H3 = 48;
    const int NVOX = in_sizes[0] / 64;

    char* wsb = (char*)d_ws;
    size_t cur = 0;
    auto alloc = [&](size_t bytes) -> char* {
        cur = (cur + 255) & ~(size_t)255;
        char* p = wsb + cur;
        cur += bytes;
        return p;
    };

    float*    x0f = (float*)alloc((size_t)Bn * G * G * 64 * 4);   // reused as y
    _Float16* x0h = (_Float16*)alloc((size_t)Bn * G * G * 64 * 2);
    float* m0 = (float*)alloc((size_t)Bn * G * G * 4);
    float* m1 = (float*)alloc((size_t)Bn * H1 * H1 * 4);
    float* m2 = (float*)alloc((size_t)Bn * H2 * H2 * 4);
    float* m3 = (float*)alloc((size_t)Bn * H3 * H3 * 4);
    _Float16* p192a = (_Float16*)alloc((size_t)Bn * H1 * H1 * 128 * 2);
    _Float16* p192b = (_Float16*)alloc((size_t)Bn * H1 * H1 * 128 * 2);
    _Float16* p96a = (_Float16*)alloc((size_t)Bn * H2 * H2 * 128 * 2);
    _Float16* p96b = (_Float16*)alloc((size_t)Bn * H2 * H2 * 128 * 2);
    _Float16* p48a = (_Float16*)alloc((size_t)Bn * H3 * H3 * 256 * 2);
    _Float16* p48b = (_Float16*)alloc((size_t)Bn * H3 * H3 * 256 * 2);
    _Float16* up1 = (_Float16*)alloc((size_t)Bn * H1 * H1 * 64 * 2);
    _Float16* up2 = (_Float16*)alloc((size_t)Bn * H1 * H1 * 64 * 2);
    _Float16* up3 = (_Float16*)alloc((size_t)Bn * H1 * H1 * 128 * 2);
    _Float16* wT1  = (_Float16*)alloc((size_t)4 * 64 * 128 * 2);
    _Float16* wTs1 = (_Float16*)alloc((size_t)3 * 9 * 128 * 128 * 2);
    _Float16* wT2  = (_Float16*)alloc((size_t)4 * 128 * 128 * 2);
    _Float16* wTs2 = (_Float16*)alloc((size_t)5 * 9 * 128 * 128 * 2);
    _Float16* wT3  = (_Float16*)alloc((size_t)4 * 128 * 256 * 2);
    _Float16* wTs3 = (_Float16*)alloc((size_t)5 * 9 * 256 * 256 * 2);
    _Float16* wTd1 = (_Float16*)alloc((size_t)128 * 64 * 2);
    _Float16* wTd2 = (_Float16*)alloc((size_t)4 * 128 * 64 * 2);
    _Float16* wTd3 = (_Float16*)alloc((size_t)16 * 256 * 128 * 2);
    float* stats = (float*)alloc(512 * 4);
    float* coef = (float*)alloc(512 * 4);
    float* cnt = (float*)alloc(16 * 4);
    (void)ws_size; (void)n_in; (void)out_size;

    float* y = x0f;  // f32 conv-output buffer (aliased; x0f done after cvt)

    auto Z = [&](float* p, long n) {
        zero_f32_kernel<<<(int)((n + 255) / 256), 256, 0, stream>>>(p, n);
    };
    auto RP = [&](const float* w, _Float16* o, int taps, int ci, int co, int dec) {
        long tot = (long)taps * ci * co;
        repack_kernel<<<(int)((tot + 255) / 256), 256, 0, stream>>>(w, o, taps, ci, co, dec);
    };
    auto CONV = [&](const _Float16* in, const _Float16* wT, float* o,
                    int Hin, int Win, int Cin, int Hg, int Wg, int Nout,
                    int KH, int KW, int st, int pad, int ups, int CoutReal) {
        int mtiles = (Bn * Hg * Wg) / 16;
        dim3 grid((mtiles + 7) / 8, Nout / 64);
        conv_wmma_kernel<<<grid, 256, 0, stream>>>(in, wT, o, Hin, Win, Cin,
                                                   Hg, Wg, Nout, KH, KW, st, pad,
                                                   ups, CoutReal, mtiles);
    };
    auto BN = [&](const float* yv, const float* mask, const float* cp,
                  const float* g, const float* bb, _Float16* o, int C, int Hg, int ms) {
        bn_stats_kernel<<<C, 256, 0, stream>>>(yv, mask, stats, Bn, Hg, Hg, C, ms);
        bn_finalize_kernel<<<1, 256, 0, stream>>>(stats, cp, g, bb, coef, C);
        long tot = (long)Bn * Hg * Hg * C;
        bn_apply_kernel<<<(int)((tot + 255) / 256), 256, 0, stream>>>(yv, mask, coef, o,
                                                                      Bn, Hg, Hg, C, ms);
    };

    // 1) scatter to dense grid + mask
    Z(x0f, (long)Bn * G * G * 64);
    Z(m0, (long)Bn * G * G);
    scatter_kernel<<<(int)(((long)NVOX * 64 + 255) / 256), 256, 0, stream>>>(vf, coors, x0f, m0, NVOX, G);
    cvt_f16_kernel<<<(int)(((long)Bn * G * G * 64 + 255) / 256), 256, 0, stream>>>(
        x0f, x0h, (long)Bn * G * G * 64);

    // 2) masks + counts
    pool_mask_kernel<<<(int)(((long)Bn * H1 * H1 + 255) / 256), 256, 0, stream>>>(m0, m1, Bn, H1, H1);
    pool_mask_kernel<<<(int)(((long)Bn * H2 * H2 + 255) / 256), 256, 0, stream>>>(m1, m2, Bn, H2, H2);
    pool_mask_kernel<<<(int)(((long)Bn * H3 * H3 + 255) / 256), 256, 0, stream>>>(m2, m3, Bn, H3, H3);
    mask_count_kernel<<<1, 256, 0, stream>>>(m1, (long)Bn * H1 * H1, 1.f, cnt + 0);
    mask_count_kernel<<<1, 256, 0, stream>>>(m2, (long)Bn * H2 * H2, 1.f, cnt + 1);
    mask_count_kernel<<<1, 256, 0, stream>>>(m3, (long)Bn * H3 * H3, 1.f, cnt + 2);
    mask_count_kernel<<<1, 256, 0, stream>>>(m2, (long)Bn * H2 * H2, 4.f, cnt + 3);
    mask_count_kernel<<<1, 256, 0, stream>>>(m3, (long)Bn * H3 * H3, 16.f, cnt + 4);

    // 3) weight repack (f32 HWIO -> f16 GEMM-B layouts)
    RP(w1, wT1, 4, 64, 128, 0);
    for (int i = 0; i < 3; ++i)
        RP(s1w + (size_t)i * 9 * 128 * 128, wTs1 + (size_t)i * 9 * 128 * 128, 9, 128, 128, 0);
    RP(w2, wT2, 4, 128, 128, 0);
    for (int i = 0; i < 5; ++i)
        RP(s2w + (size_t)i * 9 * 128 * 128, wTs2 + (size_t)i * 9 * 128 * 128, 9, 128, 128, 0);
    RP(w3, wT3, 4, 128, 256, 0);
    for (int i = 0; i < 5; ++i)
        RP(s3w + (size_t)i * 9 * 256 * 256, wTs3 + (size_t)i * 9 * 256 * 256, 9, 256, 256, 0);
    RP(d1w, wTd1, 1, 128, 64, 0);
    RP(d2w, wTd2, 4, 128, 64, 1);
    RP(d3w, wTd3, 16, 256, 128, 1);

    // 4) block 1 (384 -> 192)
    CONV(x0h, wT1, y, G, G, 64, H1, H1, 128, 2, 2, 2, 0, 1, 128);
    BN(y, m1, cnt + 0, g1, b1, p192a, 128, H1, 1);
    _Float16* curp = p192a;
    _Float16* altp = p192b;
    for (int i = 0; i < 3; ++i) {
        CONV(curp, wTs1 + (size_t)i * 9 * 128 * 128, y, H1, H1, 128, H1, H1, 128, 3, 3, 1, 1, 1, 128);
        BN(y, m1, cnt + 0, s1g + i * 128, s1b + i * 128, altp, 128, H1, 1);
        _Float16* t = curp; curp = altp; altp = t;
    }
    _Float16* x1 = curp;

    // 5) block 2 (192 -> 96)
    CONV(x1, wT2, y, H1, H1, 128, H2, H2, 128, 2, 2, 2, 0, 1, 128);
    BN(y, m2, cnt + 1, g2, b2, p96a, 128, H2, 1);
    curp = p96a; altp = p96b;
    for (int i = 0; i < 5; ++i) {
        CONV(curp, wTs2 + (size_t)i * 9 * 128 * 128, y, H2, H2, 128, H2, H2, 128, 3, 3, 1, 1, 1, 128);
        BN(y, m2, cnt + 1, s2g + i * 128, s2b + i * 128, altp, 128, H2, 1);
        _Float16* t = curp; curp = altp; altp = t;
    }
    _Float16* x2 = curp;

    // 6) block 3 (96 -> 48)
    CONV(x2, wT3, y, H2, H2, 128, H3, H3, 256, 2, 2, 2, 0, 1, 256);
    BN(y, m3, cnt + 2, g3, b3, p48a, 256, H3, 1);
    curp = p48a; altp = p48b;
    for (int i = 0; i < 5; ++i) {
        CONV(curp, wTs3 + (size_t)i * 9 * 256 * 256, y, H3, H3, 256, H3, H3, 256, 3, 3, 1, 1, 1, 256);
        BN(y, m3, cnt + 2, s3g + i * 256, s3b + i * 256, altp, 256, H3, 1);
        _Float16* t = curp; curp = altp; altp = t;
    }
    _Float16* x3 = curp;

    // 7) deconv branches back to 192
    CONV(x1, wTd1, y, H1, H1, 128, H1, H1, 64, 1, 1, 1, 0, 1, 64);
    BN(y, m1, cnt + 0, d1g, d1b, up1, 64, H1, 1);

    CONV(x2, wTd2, y, H2, H2, 128, H2, H2, 256, 1, 1, 1, 0, 2, 64);
    BN(y, m2, cnt + 3, d2g, d2b, up2, 64, H1, 2);

    CONV(x3, wTd3, y, H3, H3, 256, H3, H3, 2048, 1, 1, 1, 0, 4, 128);
    BN(y, m3, cnt + 4, d3g, d3b, up3, 128, H1, 4);

    // 8) heads
    int P = Bn * H1 * H1;
    long totH = (long)P * 22;
    heads_kernel<<<(int)((totH + 255) / 256), 256, 0, stream>>>(
        up1, up2, up3, boxw, boxb, clsw, clsb, dirw, dirb, (float*)d_out, P);
}
